// RNN_44676249813134
// MI455X (gfx1250) — compile-verified, matrix operand order checked by reading, not run on hardware
//
#include <hip/hip_runtime.h>
#include <hip/hip_bf16.h>
#include <math.h>

// ---------------- problem constants (match reference) ----------------
constexpr int VOCAB = 32000;
constexpr int D_IN  = 512;
constexpr int D_H   = 1024;
constexpr int D_OUT = 32000;
constexpr int BB    = 64;    // batch
constexpr int SS    = 512;   // sequence

// ---------------- types ----------------
typedef __bf16 bf16_t;
typedef __attribute__((ext_vector_type(16))) __bf16 v16bf;
typedef __attribute__((ext_vector_type(8)))  __bf16 v8bf;
typedef __attribute__((ext_vector_type(8)))  float  v8f;

#define DEVINL __device__ __forceinline__

// Load one wave's A- or B-tile fragment for V_WMMA_F32_16X16X32_BF16.
// CDNA5 16-bit A/B layout (wave32): lane holds row m=lane&15, half=lane>>4;
// vector elements 0..7 cover K=[8*half, 8*half+8), elements 8..15 cover
// K=[16+8*half, 16+8*half+8). Row-major bf16 (ld=K, K%8==0) => two 16-byte
// contiguous loads per fragment (global_load_b128 or ds_load_b128).
DEVINL v16bf load_frag(const bf16_t* __restrict__ p) {
  union { v16bf v; v8bf h[2]; } u;
  u.h[0] = *reinterpret_cast<const v8bf*>(p);
  u.h[1] = *reinterpret_cast<const v8bf*>(p + 16);
  return u.v;
}

DEVINL v8f wmma_bf16(v16bf a, v16bf b, v8f c) {
  return __builtin_amdgcn_wmma_f32_16x16x32_bf16(false, a, false, b,
                                                 (short)0, c, false, false);
}

// Stage `bytes` (multiple of blockDim.x*16) from global to LDS using the
// CDNA5 async-copy path: each lane moves 16B per instruction, tracked by
// ASYNCcnt. Waits (s_wait_asynccnt 0) and barriers before returning.
DEVINL void async_stage_to_lds(bf16_t* lds_dst, const bf16_t* gsrc,
                               int bytes, int tid, int nthreads) {
  const uint32_t lds_base = (uint32_t)(uintptr_t)lds_dst;  // low 32b = LDS offset
  for (int ofs = tid * 16; ofs < bytes; ofs += nthreads * 16) {
    uint32_t la = lds_base + (uint32_t)ofs;
    const char* ga = (const char*)gsrc + ofs;
    asm volatile("global_load_async_to_lds_b128 %0, %1, off"
                 :: "v"(la), "v"(ga) : "memory");
  }
  asm volatile("s_wait_asynccnt 0x0" ::: "memory");
  __syncthreads();
}

// ---------------- conversion kernels ----------------

// dst[o][k] = (bf16) src[k][o]   (src row-major [K][O], dst row-major [O][K])
__global__ __launch_bounds__(256)
void k_cvt_transpose(const float* __restrict__ src, bf16_t* __restrict__ dst,
                     int K, int O) {
  long idx = (long)blockIdx.x * blockDim.x + threadIdx.x;
  long total = (long)K * O;
  if (idx >= total) return;
  long o = idx / K;
  long k = idx - o * K;
  dst[idx] = (bf16_t)src[k * O + o];
}

__global__ __launch_bounds__(256)
void k_cvt_f32_bf16(const float* __restrict__ src, bf16_t* __restrict__ dst,
                    long n) {
  long idx = (long)blockIdx.x * blockDim.x + threadIdx.x;
  if (idx < n) dst[idx] = (bf16_t)src[idx];
}

// X[(s*B + b)*D_IN + i] = (bf16) embed[tokens[b*S + s]][i]
__global__ __launch_bounds__(256)
void k_gather_embed(const int* __restrict__ tokens,
                    const float* __restrict__ embed,
                    bf16_t* __restrict__ X) {
  long idx = (long)blockIdx.x * blockDim.x + threadIdx.x;
  long total = (long)SS * BB * D_IN;
  if (idx >= total) return;
  int  i = (int)(idx % D_IN);
  long t = idx / D_IN;           // t = s*B + b
  int  b = (int)(t % BB);
  int  s = (int)(t / BB);
  int  tok = tokens[(long)b * SS + s];
  X[idx] = (bf16_t)embed[(long)tok * D_IN + i];
}

// ---------------- WMMA GEMM: C[M][N] = A[M][K] * Bt[N][K]^T + bias[N] ----------------
// A, Bt bf16 row-major (ld = K), C f32 row-major (ld = N).
// Block = 128 threads (4 waves); all 4 waves share mtile = blockIdx.y, so the
// 16xK A panel is async-staged into LDS once and fragments come from ds_load.
// Wave computes a 16(M) x 64(N) strip: 4 f32 accumulators, A fragment reused
// across 4 WMMAs per 32-wide K step. grid.x covers N in 256-wide chunks
// (launches are exact: every block has valid ncol), grid.y = M/16.
// Dynamic LDS: 16*K*2 bytes.
__global__ __launch_bounds__(128)
void k_gemm_bf16(const bf16_t* __restrict__ A,
                 const bf16_t* __restrict__ Bt,
                 const float*  __restrict__ bias,
                 float*        __restrict__ C,
                 int N, int K) {
  extern __shared__ __align__(16) char smem_raw[];
  bf16_t* sA = (bf16_t*)smem_raw;                 // [16][K]

  const int lane = threadIdx.x & 31;   // wave32
  const int wave = threadIdx.x >> 5;
  const int nl   = lane & 15;
  const int half = lane >> 4;
  const int mtile = blockIdx.y;
  const int ncol  = blockIdx.x * 256 + wave * 64;

  // stage the shared 16-row A panel into LDS (async copy, ASYNCcnt)
  async_stage_to_lds(sA, A + (size_t)mtile * 16 * K,
                     16 * K * (int)sizeof(bf16_t), threadIdx.x, 128);

  const bf16_t* pa  = sA + (size_t)nl * K + 8 * half;         // LDS
  const bf16_t* pb0 = Bt + (size_t)(ncol +  0 + nl) * K + 8 * half;
  const bf16_t* pb1 = Bt + (size_t)(ncol + 16 + nl) * K + 8 * half;
  const bf16_t* pb2 = Bt + (size_t)(ncol + 32 + nl) * K + 8 * half;
  const bf16_t* pb3 = Bt + (size_t)(ncol + 48 + nl) * K + 8 * half;

  v8f acc0 = {}, acc1 = {}, acc2 = {}, acc3 = {};
  for (int k = 0; k < K; k += 32) {
    __builtin_prefetch(pb0 + k + 64, 0, 3);      // prefetch streaming B panels
    __builtin_prefetch(pb2 + k + 64, 0, 3);
    v16bf a  = load_frag(pa  + k);               // ds_load_b128 x2
    v16bf b0 = load_frag(pb0 + k);               // global_load_b128 x2 each
    v16bf b1 = load_frag(pb1 + k);
    v16bf b2 = load_frag(pb2 + k);
    v16bf b3 = load_frag(pb3 + k);
    acc0 = wmma_bf16(a, b0, acc0);
    acc1 = wmma_bf16(a, b1, acc1);
    acc2 = wmma_bf16(a, b2, acc2);
    acc3 = wmma_bf16(a, b3, acc3);
  }

  // C/D layout: element r of lane holds (m = 8*half + r, n = lane&15)
  const int row0 = mtile * 16 + 8 * half;
  const float bia0 = bias[ncol +  0 + nl];
  const float bia1 = bias[ncol + 16 + nl];
  const float bia2 = bias[ncol + 32 + nl];
  const float bia3 = bias[ncol + 48 + nl];
#pragma unroll
  for (int r = 0; r < 8; ++r) {
    float* crow = C + (size_t)(row0 + r) * N + ncol + nl;
    crow[ 0] = acc0[r] + bia0;
    crow[16] = acc1[r] + bia1;
    crow[32] = acc2[r] + bia2;
    crow[48] = acc3[r] + bia3;
  }
}

// ---------------- one RNN step: h_next = tanh(xw_s + h_prev @ W_hh) ----------------
// M=64, N=K=1024. 64 strips of 16x64 -> 16 blocks x 4 waves. All 4 waves in a
// block share mtile (strip>>4), so the 16x1024 h_prev panel (32KB) is
// async-staged to LDS once per block per step.
__global__ __launch_bounds__(128)
void k_rnn_step(const bf16_t* __restrict__ hprev,   // [64][1024] bf16
                const bf16_t* __restrict__ Whh_t,   // [1024][1024] bf16 (transposed)
                const float*  __restrict__ xw_s,    // [64][1024] f32
                bf16_t*       __restrict__ hnext) { // [64][1024] bf16
  constexpr int N = D_H, K = D_H;
  __shared__ __align__(16) bf16_t sA[16 * D_H];     // 32 KB

  const int lane = threadIdx.x & 31;
  const int wave = threadIdx.x >> 5;
  const int strip = blockIdx.x * 4 + wave;   // 0..63
  const int mtile = strip >> 4;              // 0..3  (same for all waves in block)
  const int ncol  = (strip & 15) * 64;       // 0..960
  const int nl    = lane & 15;
  const int half  = lane >> 4;

  async_stage_to_lds(sA, hprev + (size_t)mtile * 16 * K,
                     16 * K * (int)sizeof(bf16_t), threadIdx.x, 128);

  const bf16_t* pa  = sA + (size_t)nl * K + 8 * half;          // LDS
  const bf16_t* pb0 = Whh_t + (size_t)(ncol +  0 + nl) * K + 8 * half;
  const bf16_t* pb1 = Whh_t + (size_t)(ncol + 16 + nl) * K + 8 * half;
  const bf16_t* pb2 = Whh_t + (size_t)(ncol + 32 + nl) * K + 8 * half;
  const bf16_t* pb3 = Whh_t + (size_t)(ncol + 48 + nl) * K + 8 * half;

  v8f acc0 = {}, acc1 = {}, acc2 = {}, acc3 = {};
#pragma unroll 4
  for (int k = 0; k < K; k += 32) {
    v16bf a  = load_frag(pa  + k);
    v16bf b0 = load_frag(pb0 + k);
    v16bf b1 = load_frag(pb1 + k);
    v16bf b2 = load_frag(pb2 + k);
    v16bf b3 = load_frag(pb3 + k);
    acc0 = wmma_bf16(a, b0, acc0);
    acc1 = wmma_bf16(a, b1, acc1);
    acc2 = wmma_bf16(a, b2, acc2);
    acc3 = wmma_bf16(a, b3, acc3);
  }

  const int row0 = mtile * 16 + 8 * half;
#pragma unroll
  for (int r = 0; r < 8; ++r) {
    const size_t base = (size_t)(row0 + r) * N + ncol + nl;
    hnext[base +  0] = (bf16_t)tanhf(acc0[r] + xw_s[base +  0]);
    hnext[base + 16] = (bf16_t)tanhf(acc1[r] + xw_s[base + 16]);
    hnext[base + 32] = (bf16_t)tanhf(acc2[r] + xw_s[base + 32]);
    hnext[base + 48] = (bf16_t)tanhf(acc3[r] + xw_s[base + 48]);
  }
}

// ---------------- host orchestration ----------------
extern "C" void kernel_launch(void* const* d_in, const int* in_sizes, int n_in,
                              void* d_out, int out_size, void* d_ws, size_t ws_size,
                              hipStream_t stream) {
  const int*   tokens  = (const int*)  d_in[0];   // [B, S]
  const float* h0      = (const float*)d_in[1];   // [B, D_H]
  const float* embed   = (const float*)d_in[2];   // [VOCAB, D_IN]
  const float* W_ih    = (const float*)d_in[3];   // [D_IN, D_H]
  const float* W_hh    = (const float*)d_in[4];   // [D_H, D_H]
  const float* bias_hh = (const float*)d_in[5];   // [D_H]
  const float* W_out   = (const float*)d_in[6];   // [D_H, D_OUT]
  const float* b_out   = (const float*)d_in[7];   // [D_OUT]
  float* out = (float*)d_out;                     // [B, D_OUT]

  // workspace carve-out (256B-aligned slices)
  size_t off = 0;
  auto take = [&](size_t bytes) -> char* {
    char* p = (char*)d_ws + off;
    off += (bytes + 255) & ~(size_t)255;
    return p;
  };
  bf16_t* wih_t  = (bf16_t*)take((size_t)D_H   * D_IN * sizeof(bf16_t)); // [D_H][D_IN]
  bf16_t* whh_t  = (bf16_t*)take((size_t)D_H   * D_H  * sizeof(bf16_t)); // [D_H][D_H]
  bf16_t* wout_t = (bf16_t*)take((size_t)D_OUT * D_H  * sizeof(bf16_t)); // [D_OUT][D_H]
  bf16_t* xbf    = (bf16_t*)take((size_t)SS * BB * D_IN * sizeof(bf16_t)); // [S*B][D_IN]
  float*  xw     = (float*) take((size_t)SS * BB * D_H  * sizeof(float));  // [S][B][D_H]
  bf16_t* hbuf0  = (bf16_t*)take((size_t)BB * D_H * sizeof(bf16_t));
  bf16_t* hbuf1  = (bf16_t*)take((size_t)BB * D_H * sizeof(bf16_t));
  bf16_t* hb[2]  = { hbuf0, hbuf1 };

  // 1) weight conversion + transpose into bf16 [out][in]
  {
    long n = (long)D_H * D_IN;
    k_cvt_transpose<<<(int)((n + 255) / 256), 256, 0, stream>>>(W_ih, wih_t, D_IN, D_H);
  }
  {
    long n = (long)D_H * D_H;
    k_cvt_transpose<<<(int)((n + 255) / 256), 256, 0, stream>>>(W_hh, whh_t, D_H, D_H);
  }
  {
    long n = (long)D_OUT * D_H;
    k_cvt_transpose<<<(int)((n + 255) / 256), 256, 0, stream>>>(W_out, wout_t, D_H, D_OUT);
  }
  // 2) h0 -> bf16
  {
    long n = (long)BB * D_H;
    k_cvt_f32_bf16<<<(int)((n + 255) / 256), 256, 0, stream>>>(h0, hbuf0, n);
  }
  // 3) embedding gather -> bf16 X [S*B][D_IN]
  {
    long n = (long)SS * BB * D_IN;
    k_gather_embed<<<(int)((n + 255) / 256), 256, 0, stream>>>(tokens, embed, xbf);
  }
  // 4) input projection: XW = X @ W_ih + bias_hh   (M=32768, N=1024, K=512)
  {
    dim3 grid(D_H / 256, (SS * BB) / 16);
    size_t shmem = (size_t)16 * D_IN * sizeof(bf16_t);   // 16KB A panel
    k_gemm_bf16<<<grid, 128, shmem, stream>>>(xbf, wih_t, bias_hh, xw, D_H, D_IN);
  }
  // 5) 512 sequential recurrence steps (stream order provides the scan dependency)
  for (int s = 0; s < SS; ++s) {
    k_rnn_step<<<16, 128, 0, stream>>>(hb[s & 1], whh_t,
                                       xw + (size_t)s * BB * D_H,
                                       hb[(s + 1) & 1]);
  }
  // after 512 steps h_last lives in hbuf0 (512 is even)
  // 6) output projection: out = h_last @ W_out + b_out   (M=64, N=32000, K=1024)
  {
    dim3 grid(D_OUT / 256, BB / 16);
    size_t shmem = (size_t)16 * D_H * sizeof(bf16_t);    // 32KB A panel
    k_gemm_bf16<<<grid, 128, shmem, stream>>>(hb[0], wout_t, b_out, out, D_OUT, D_H);
  }
}
